// DR2FWL2Conv_39848706572456
// MI455X (gfx1250) — compile-verified, matrix operand order checked by reading, not run on hardware
//
#include <hip/hip_runtime.h>
#include <hip/hip_bf16.h>

// DR2FWL2Conv for MI455X (gfx1250, wave32, WMMA f32_16x16x32_f16).
//
// Workspace layout (floats): a0[N*64] | a1[E1*64] | a2[E2*64] | tmp[E2*64]
// total = 64*(N+E1+E2+E2)*4 bytes ~= 525 MB.

typedef __attribute__((ext_vector_type(16))) _Float16 v16h;
typedef __attribute__((ext_vector_type(8)))  _Float16 v8h;
typedef __attribute__((ext_vector_type(2)))  _Float16 v2h;
typedef __attribute__((ext_vector_type(8)))  float    v8f;

// Load a 16x32 f16 WMMA operand fragment from a row-major (stride 64 halves)
// LDS buffer. CDNA5 ISA layout: lanes 0-15 hold K=kc..kc+7 (VGPR0-3) and
// K=kc+16..kc+23 (VGPR4-7); lanes 16-31 the same +8.
__device__ __forceinline__ v16h load_frag16(const _Float16* __restrict__ base,
                                            int row, int kc, int hi) {
    const _Float16* p = base + row * 64 + kc + hi * 8;
    v8h lo = *(const v8h*)(p);
    v8h hq = *(const v8h*)(p + 16);
    v16h r;
#pragma unroll
    for (int h = 0; h < 8; ++h) { r[h] = lo[h]; r[h + 8] = hq[h]; }
    return r;
}

// 16x64 tile (sA) times 64x64 weight whose 8 B-fragments are pre-hoisted in
// registers (bw[kc2*4+nt]). Per kc-chunk: 2 ds_load_b128 for A, then 4
// back-to-back WMMAs.
__device__ __forceinline__ void gemm16x64_regB(const _Float16* __restrict__ sA,
                                               const v16h bw[8],
                                               v8f acc[4], int m, int hi) {
#pragma unroll
    for (int kc2 = 0; kc2 < 2; ++kc2) {
        v16h a = load_frag16(sA, m, kc2 * 32, hi);
#pragma unroll
        for (int nt = 0; nt < 4; ++nt) {
            acc[nt] = __builtin_amdgcn_wmma_f32_16x16x32_f16(
                false, a, false, bw[kc2 * 4 + nt], (short)0, acc[nt], false, false);
        }
    }
}

// LDS-resident weight version (used for the per-tile second weight in
// DUAL/SECOND paths to cap register pressure).
__device__ __forceinline__ void gemm16x64(const _Float16* __restrict__ sA,
                                          const _Float16* __restrict__ sWt,
                                          v8f acc[4], int m, int hi) {
#pragma unroll
    for (int kc = 0; kc < 64; kc += 32) {
        v16h a = load_frag16(sA, m, kc, hi);
#pragma unroll
        for (int nt = 0; nt < 4; ++nt) {
            v16h b = load_frag16(sWt, nt * 16 + m, kc, hi);
            acc[nt] = __builtin_amdgcn_wmma_f32_16x16x32_f16(
                false, a, false, b, (short)0, acc[nt], false, false);
        }
    }
}

// Stage 16 rows (gather / gather-pair-sum) fp32 -> f16 into the wave's LDS
// tile. One coalesced 256B row per iteration; packed 2xf16 LDS stores.
__device__ __forceinline__ void stage_tile(_Float16* __restrict__ sA,
                                           const float* __restrict__ X1,
                                           const int* __restrict__ i1,
                                           const float* __restrict__ X2,
                                           const int* __restrict__ i2,
                                           long r0, long nrows, int lane) {
#pragma unroll 4
    for (int rr = 0; rr < 16; ++rr) {
        long r = r0 + rr;
        float vx = 0.0f, vy = 0.0f;
        if (r < nrows) {
            long g1 = i1 ? (long)i1[r] : r;
            const float* p1 = X1 + g1 * 64 + lane * 2;
            vx = p1[0];
            vy = p1[1];
            if (X2) {
                long g2 = i2 ? (long)i2[r] : r;
                const float* p2 = X2 + g2 * 64 + lane * 2;
                vx += p2[0];
                vy += p2[1];
            }
        }
        v2h pk;
        pk[0] = (_Float16)vx;
        pk[1] = (_Float16)vy;
        *(v2h*)(sA + rr * 64 + lane * 2) = pk;
    }
}

// One kernel for all phases (persistent blocks, grid-stride over tiles):
//   y = [lin2]( relu1( lin1a(X1[i1] (+ X2[i2])) [+ lin1b(X2[i2]) in DUAL] ) )
//   OUT[row] += y                     (ATOMIC=false, aligned)
//   atomicAdd(OUT[oidx[row]], y)      (ATOMIC=true, segment_sum scatter)
template <bool DUAL, bool RELU1, bool SECOND, bool ATOMIC>
__global__ __launch_bounds__(256) void frag_mlp_kernel(
    const float* __restrict__ X1, const int* __restrict__ i1,
    const float* __restrict__ X2, const int* __restrict__ i2,
    const float* __restrict__ W1a, const float* __restrict__ b1a,
    const float* __restrict__ W1b, const float* __restrict__ b1b,
    const float* __restrict__ W2,  const float* __restrict__ b2,
    float* __restrict__ OUT, const int* __restrict__ oidx, long nrows) {
    __shared__ __align__(16) _Float16 sW1[4096];
    __shared__ __align__(16) _Float16 sW2[4096];
    __shared__ float sB1[64];
    __shared__ float sB2[64];
    __shared__ __align__(16) _Float16 sA[8][1024];

    const int tid  = threadIdx.x;
    const int wave = tid >> 5;
    const int lane = tid & 31;
    const int m    = lane & 15;
    const int hi   = lane >> 4;

    // Stage transposed f16 weights into LDS once per (persistent) block.
    // Global reads coalesced; transpose absorbed by LDS writes.
    for (int t = tid; t < 4096; t += 256) {
        int k = t >> 6, n = t & 63;
        sW1[n * 64 + k] = (_Float16)W1a[t];
    }
    if (DUAL || SECOND) {
        const float* Wx = DUAL ? W1b : W2;
        for (int t = tid; t < 4096; t += 256) {
            int k = t >> 6, n = t & 63;
            sW2[n * 64 + k] = (_Float16)Wx[t];
        }
    }
    if (tid < 64) {
        float bb = b1a[tid];
        if (DUAL) bb += b1b[tid];
        sB1[tid] = bb;
        sB2[tid] = SECOND ? b2[tid] : 0.0f;
    }
    __syncthreads();
    // From here on each wave touches only its private sA[wave] slice;
    // same-wave LDS ops are in-order (DScnt), so no block barriers needed.

    _Float16* myA = sA[wave];

    // Hoist the loop-invariant W1 B-fragments into registers (8 x v16h).
    v16h bw1[8];
#pragma unroll
    for (int kc2 = 0; kc2 < 2; ++kc2)
#pragma unroll
        for (int nt = 0; nt < 4; ++nt)
            bw1[kc2 * 4 + nt] = load_frag16(sW1, nt * 16 + m, kc2 * 32, hi);

    float bn1[4];
    float bn2[4];
#pragma unroll
    for (int nt = 0; nt < 4; ++nt) {
        bn1[nt] = sB1[nt * 16 + m];
        bn2[nt] = sB2[nt * 16 + m];
    }

    const long ntiles = (nrows + 127) >> 7;  // 128 rows per block-iteration
    for (long tb = blockIdx.x; tb < ntiles; tb += gridDim.x) {
        const long r0 = tb * 128 + (long)wave * 16;

        v8f acc[4] = {};

        stage_tile(myA, X1, i1, DUAL ? nullptr : X2, DUAL ? nullptr : i2,
                   r0, nrows, lane);
        gemm16x64_regB(myA, bw1, acc, m, hi);

        if (DUAL) {
            stage_tile(myA, X2, i2, nullptr, nullptr, r0, nrows, lane);
            gemm16x64(myA, sW2, acc, m, hi);
        }

        if (SECOND) {
            // bias + relu, restage h into LDS in A-layout (C layout: lane->N,
            // vgpr r -> M = r + hi*8), then second GEMM.
#pragma unroll
            for (int nt = 0; nt < 4; ++nt) {
#pragma unroll
                for (int r = 0; r < 8; ++r) {
                    int mm = r + hi * 8;
                    float v = acc[nt][r] + bn1[nt];
                    if (RELU1) v = fmaxf(v, 0.0f);
                    myA[mm * 64 + nt * 16 + m] = (_Float16)v;
                }
            }
            v8f acc2[4] = {};
            gemm16x64(myA, sW2, acc2, m, hi);
#pragma unroll
            for (int r = 0; r < 8; ++r) {
                int mm = r + hi * 8;
                long row = r0 + mm;
                if (row < nrows) {
                    long obase = ATOMIC ? (long)oidx[row] * 64 : row * 64;
#pragma unroll
                    for (int nt = 0; nt < 4; ++nt) {
                        float v = acc2[nt][r] + bn2[nt];
                        int col = nt * 16 + m;
                        if (ATOMIC) atomicAdd(OUT + obase + col, v);
                        else        OUT[obase + col] += v;
                    }
                }
            }
        } else {
#pragma unroll
            for (int r = 0; r < 8; ++r) {
                int mm = r + hi * 8;
                long row = r0 + mm;
                if (row < nrows) {
                    long obase = ATOMIC ? (long)oidx[row] * 64 : row * 64;
#pragma unroll
                    for (int nt = 0; nt < 4; ++nt) {
                        float v = acc[nt][r] + bn1[nt];
                        if (RELU1) v = fmaxf(v, 0.0f);
                        int col = nt * 16 + m;
                        if (ATOMIC) atomicAdd(OUT + obase + col, v);
                        else        OUT[obase + col] += v;
                    }
                }
            }
        }
    }
}

static inline int grid_for(long rows) {
    long t = (rows + 127) / 128;
    if (t > 2048) t = 2048;  // persistent blocks; grid-stride covers the rest
    return (int)t;
}

extern "C" void kernel_launch(void* const* d_in, const int* in_sizes, int n_in,
                              void* d_out, int out_size, void* d_ws, size_t ws_size,
                              hipStream_t stream) {
    (void)n_in; (void)out_size; (void)ws_size;
    const float* x0  = (const float*)d_in[0];
    const float* x1  = (const float*)d_in[1];
    const float* x2  = (const float*)d_in[2];
    const float* Ws  = (const float*)d_in[3];
    const float* bsv = (const float*)d_in[4];
    const int* ei1    = (const int*)d_in[5];
    const int* ei2    = (const int*)d_in[6];
    const int* tri111 = (const int*)d_in[7];
    const int* tri222 = (const int*)d_in[8];
    const int* tri112 = (const int*)d_in[9];
    const int* tri221 = (const int*)d_in[10];
    const int* inv1   = (const int*)d_in[11];
    const int* inv2   = (const int*)d_in[12];

    const long N  = in_sizes[0] / 64;
    const long E1 = in_sizes[1] / 64;
    const long E2 = in_sizes[2] / 64;
    const long T  = in_sizes[7] / 3;

    float* a0  = (float*)d_ws;
    float* a1  = a0 + N * 64;
    float* a2  = a1 + E1 * 64;
    float* tmp = a2 + E2 * 64;

    auto W = [&](int i) { return Ws + (size_t)i * 4096; };
    auto B = [&](int i) { return bsv + (size_t)i * 64; };

    hipMemsetAsync(a0, 0, (size_t)N * 64 * 4, stream);
    hipMemsetAsync(a1, 0, (size_t)E1 * 64 * 4, stream);
    hipMemsetAsync(a2, 0, (size_t)E2 * 64 * 4, stream);

    // ---- aggr (0,1,1) ----
    frag_mlp_kernel<false, true, true, false><<<grid_for(E1), 256, 0, stream>>>(
        x0, ei1, x0, ei1 + E1, W(0), B(0), nullptr, nullptr, W(2), B(2),
        a1, nullptr, E1);
    hipMemsetAsync(tmp, 0, (size_t)N * 64 * 4, stream);
    frag_mlp_kernel<false, true, false, true><<<grid_for(E1), 256, 0, stream>>>(
        x1, nullptr, nullptr, nullptr, W(1), B(1), nullptr, nullptr, nullptr, nullptr,
        tmp, ei1, E1);
    frag_mlp_kernel<false, false, false, false><<<grid_for(N), 256, 0, stream>>>(
        tmp, nullptr, nullptr, nullptr, W(3), B(3), nullptr, nullptr, nullptr, nullptr,
        a0, nullptr, N);

    // ---- aggr (0,2,2) ----
    frag_mlp_kernel<false, true, true, false><<<grid_for(E2), 256, 0, stream>>>(
        x0, ei2, x0, ei2 + E2, W(4), B(4), nullptr, nullptr, W(6), B(6),
        a2, nullptr, E2);
    hipMemsetAsync(tmp, 0, (size_t)N * 64 * 4, stream);
    frag_mlp_kernel<false, true, false, true><<<grid_for(E2), 256, 0, stream>>>(
        x2, nullptr, nullptr, nullptr, W(5), B(5), nullptr, nullptr, nullptr, nullptr,
        tmp, ei2, E2);
    frag_mlp_kernel<false, false, false, false><<<grid_for(N), 256, 0, stream>>>(
        tmp, nullptr, nullptr, nullptr, W(7), B(7), nullptr, nullptr, nullptr, nullptr,
        a0, nullptr, N);

    // ---- aggr (1,1,1) ----
    hipMemsetAsync(tmp, 0, (size_t)E1 * 64 * 4, stream);
    frag_mlp_kernel<false, true, false, true><<<grid_for(T), 256, 0, stream>>>(
        x1, tri111 + T, x1, tri111 + 2 * T, W(8), B(8), nullptr, nullptr, nullptr, nullptr,
        tmp, tri111, T);
    frag_mlp_kernel<false, false, false, false><<<grid_for(E1), 256, 0, stream>>>(
        tmp, nullptr, nullptr, nullptr, W(9), B(9), nullptr, nullptr, nullptr, nullptr,
        a1, nullptr, E1);

    // ---- aggr (2,2,2) ----
    hipMemsetAsync(tmp, 0, (size_t)E2 * 64 * 4, stream);
    frag_mlp_kernel<false, true, false, true><<<grid_for(T), 256, 0, stream>>>(
        x2, tri222 + T, x2, tri222 + 2 * T, W(10), B(10), nullptr, nullptr, nullptr, nullptr,
        tmp, tri222, T);
    frag_mlp_kernel<false, false, false, false><<<grid_for(E2), 256, 0, stream>>>(
        tmp, nullptr, nullptr, nullptr, W(11), B(11), nullptr, nullptr, nullptr, nullptr,
        a2, nullptr, E2);

    // ---- aggr (1,1,2) ----
    hipMemsetAsync(tmp, 0, (size_t)E2 * 64 * 4, stream);
    frag_mlp_kernel<false, true, false, true><<<grid_for(T), 256, 0, stream>>>(
        x1, tri112, x1, tri112 + T, W(12), B(12), nullptr, nullptr, nullptr, nullptr,
        tmp, tri112 + 2 * T, T);
    frag_mlp_kernel<false, false, false, false><<<grid_for(E2), 256, 0, stream>>>(
        tmp, nullptr, nullptr, nullptr, W(15), B(15), nullptr, nullptr, nullptr, nullptr,
        a2, nullptr, E2);
    hipMemsetAsync(tmp, 0, (size_t)E1 * 64 * 4, stream);
    frag_mlp_kernel<true, true, false, true><<<grid_for(T), 256, 0, stream>>>(
        x1, tri112 + T, x2, tri112 + 2 * T, W(13), B(13), W(14), B(14), nullptr, nullptr,
        tmp, tri112, T);
    frag_mlp_kernel<false, false, false, false><<<grid_for(E1), 256, 0, stream>>>(
        tmp, nullptr, tmp, inv1, W(16), B(16), nullptr, nullptr, nullptr, nullptr,
        a1, nullptr, E1);

    // ---- aggr (2,2,1) ----
    hipMemsetAsync(tmp, 0, (size_t)E1 * 64 * 4, stream);
    frag_mlp_kernel<false, true, false, true><<<grid_for(T), 256, 0, stream>>>(
        x2, tri221, x2, tri221 + T, W(17), B(17), nullptr, nullptr, nullptr, nullptr,
        tmp, tri221 + 2 * T, T);
    frag_mlp_kernel<false, false, false, false><<<grid_for(E1), 256, 0, stream>>>(
        tmp, nullptr, nullptr, nullptr, W(20), B(20), nullptr, nullptr, nullptr, nullptr,
        a1, nullptr, E1);
    hipMemsetAsync(tmp, 0, (size_t)E2 * 64 * 4, stream);
    frag_mlp_kernel<true, true, false, true><<<grid_for(T), 256, 0, stream>>>(
        x2, tri221 + T, x1, tri221 + 2 * T, W(18), B(18), W(19), B(19), nullptr, nullptr,
        tmp, tri221, T);
    frag_mlp_kernel<false, false, false, false><<<grid_for(E2), 256, 0, stream>>>(
        tmp, nullptr, tmp, inv2, W(21), B(21), nullptr, nullptr, nullptr, nullptr,
        a2, nullptr, E2);

    // ---- output MLPs with residual ----
    float* out0 = (float*)d_out;
    float* out1 = out0 + N * 64;
    float* out2 = out1 + E1 * 64;
    hipMemcpyAsync(out0, x0, (size_t)N * 64 * 4, hipMemcpyDeviceToDevice, stream);
    hipMemcpyAsync(out1, x1, (size_t)E1 * 64 * 4, hipMemcpyDeviceToDevice, stream);
    hipMemcpyAsync(out2, x2, (size_t)E2 * 64 * 4, hipMemcpyDeviceToDevice, stream);

    frag_mlp_kernel<false, true, true, false><<<grid_for(N), 256, 0, stream>>>(
        x0, nullptr, a0, nullptr, W(22), B(22), nullptr, nullptr, W(23), B(23),
        out0, nullptr, N);
    frag_mlp_kernel<false, true, true, false><<<grid_for(E1), 256, 0, stream>>>(
        x1, nullptr, a1, nullptr, W(24), B(24), nullptr, nullptr, W(25), B(25),
        out1, nullptr, E1);
    frag_mlp_kernel<false, true, true, false><<<grid_for(E2), 256, 0, stream>>>(
        x2, nullptr, a2, nullptr, W(26), B(26), nullptr, nullptr, W(27), B(27),
        out2, nullptr, E2);
}
